// MambaBlock_7799660609679
// MI455X (gfx1250) — matrix operand order repaired and checked by reading, not compiled, not run on hardware
//
#include <hip/hip_runtime.h>

typedef __attribute__((ext_vector_type(16))) __bf16 v16bf;
typedef __attribute__((ext_vector_type(8)))  __bf16 v8bf;
typedef __attribute__((ext_vector_type(8)))  float  v8f;
typedef __attribute__((ext_vector_type(4)))  float  v4f;

#define MODE_PLAIN    0
#define MODE_XP_SILU  1
#define MODE_DBC      2
#define MODE_SOFTPLUS 3

#define USE_ASYNC_LDS 1

static __device__ __forceinline__ unsigned short f32_to_bf16(float f) {
    unsigned int u = __float_as_uint(f);
    unsigned int r = (u + 0x7FFFu + ((u >> 16) & 1u)) >> 16;  // round-nearest-even
    return (unsigned short)r;
}

__global__ __launch_bounds__(256) void convert_f32_bf16(
    const float* __restrict__ src, unsigned short* __restrict__ dst, int n)
{
    int i = blockIdx.x * blockDim.x + threadIdx.x;
    if (i < n) dst[i] = f32_to_bf16(src[i]);
}

static __device__ __forceinline__ void load_a_frag(const __bf16* p, v16bf& a) {
    *(v8bf*)&a       = *(const v8bf*)(p);        // K = kb .. kb+7
    *((v8bf*)&a + 1) = *(const v8bf*)(p + 8);    // K = kb+8 .. kb+15
}

// Y(M,N) = X(M,K)bf16 @ W(N,K)bf16^T + bias, fused epilogue.
// One wave32 -> 16(M) x 64(N) tile. Register double-buffered K-loop (step 64 =
// 2 WMMA groups): loads for buffer-1 are in flight during WMMAs on buffer-0,
// so s_wait_loadcnt never covers the just-issued loads.
template <int MODE>
__global__ __launch_bounds__(256) void gemm_wmma_bf16(
    const unsigned short* __restrict__ Xr, int ldx,
    const unsigned short* __restrict__ Wr, int ldw,
    const float* __restrict__ bias,
    int M, int N, int K,
    float* __restrict__ out0,
    float* __restrict__ out1,
    unsigned short* __restrict__ outbf)
{
    const __bf16* X = reinterpret_cast<const __bf16*>(Xr);
    const __bf16* W = reinterpret_cast<const __bf16*>(Wr);

    const int lane   = threadIdx.x & 31;
    const int wave   = (blockIdx.x * blockDim.x + threadIdx.x) >> 5;
    const int nTiles = N >> 6;
    const int m0 = (wave / nTiles) << 4;
    const int n0 = (wave % nTiles) << 6;

    v8f acc[4] = {};

    const int aKoff = (lane >> 4) << 4;   // 0 or 16
    const int bKoff = (lane >> 4) << 4;   // 0 or 16
    const __bf16* aBase = X + (size_t)(m0 + (lane & 15)) * ldx + aKoff;
    const __bf16* bPtr[4];
#pragma unroll
    for (int t = 0; t < 4; ++t)
        bPtr[t] = W + (size_t)(n0 + (t << 4) + (lane & 15)) * ldw + bKoff;

    auto load_b4 = [&](int kk, v16bf b[4]) {
#pragma unroll
        for (int t = 0; t < 4; ++t)
            b[t] = *(const v16bf*)(bPtr[t] + kk);  // 16 contiguous K per lane
    };

    v16bf a0, a1, b0[4], b1[4];
    load_a_frag(aBase, a0);
    load_b4(0, b0);

    for (int kk = 0; kk < K; kk += 64) {
        // stage next half-step while buffer-0 WMMAs execute
        load_a_frag(aBase + kk + 32, a1);
        load_b4(kk + 32, b1);
        if (kk + 128 < K) __builtin_prefetch(aBase + kk + 128, 0, 1);
#pragma unroll
        for (int t = 0; t < 4; ++t)
            acc[t] = __builtin_amdgcn_wmma_f32_16x16x32_bf16(
                false, a0, false, b0[t], (short)0, acc[t], false, false);
        if (kk + 64 < K) {
            load_a_frag(aBase + kk + 64, a0);
            load_b4(kk + 64, b0);
        }
#pragma unroll
        for (int t = 0; t < 4; ++t)
            acc[t] = __builtin_amdgcn_wmma_f32_16x16x32_bf16(
                false, a1, false, b1[t], (short)0, acc[t], false, false);
    }

    // C/D layout: lane l -> col (l&15), rows m0 + 8*(l>>4) + v (v = VGPR 0..7)
    const int rowBase = m0 + ((lane >> 4) << 3);
#pragma unroll
    for (int t = 0; t < 4; ++t) {
        const int col = n0 + (t << 4) + (lane & 15);
        const float bv = bias ? bias[col] : 0.0f;
#pragma unroll
        for (int v = 0; v < 8; ++v) {
            float z = acc[t][v] + bv;
            const size_t idx = (size_t)(rowBase + v) * N + col;
            if (MODE == MODE_PLAIN) {
                out0[idx] = z;
            } else if (MODE == MODE_XP_SILU) {
                out0[idx] = z;                                   // xp (conv input)
                out1[idx] = z / (1.0f + __expf(-z));             // x_two = silu(xp)
            } else if (MODE == MODE_DBC) {
                out0[idx] = z;                                   // dbc f32 (B,C cols)
                if (col < 64)                                    // delta_raw -> bf16
                    outbf[(size_t)(rowBase + v) * 64 + col] = f32_to_bf16(z);
            } else {                                             // softplus
                out0[idx] = (z > 20.0f) ? z : log1pf(__expf(z));
            }
        }
    }
}

// conv over feature axis, seq axis (L=64) as channels, k=3, pad 1; fused silu.
// block: 256 threads over d-chunk for fixed (b, lo); weights (192 f32) staged
// into LDS via CDNA5 async global->LDS copies (48 lanes x b128), ASYNCcnt wait.
__global__ __launch_bounds__(256) void conv_silu_kernel(
    const float* __restrict__ xp, const float* __restrict__ cw,
    const float* __restrict__ cb, float* __restrict__ x_one,
    unsigned short* __restrict__ x_one_bf)
{
    __shared__ float w[192];
    const int bo = blockIdx.x;          // b*64 + lo
    const int b  = bo >> 6;
    const int lo = bo & 63;
    const int d  = blockIdx.y * 256 + threadIdx.x;

#if USE_ASYNC_LDS
    if (threadIdx.x < 48) {
        unsigned lds_off = (unsigned)(size_t)(&w[threadIdx.x * 4]);
        const float* g = cw + (size_t)lo * 192 + threadIdx.x * 4;
        asm volatile("global_load_async_to_lds_b128 %0, %1, off"
                     :: "v"(lds_off), "v"(g) : "memory");
    }
    asm volatile("s_wait_asynccnt 0x0" ::: "memory");
    __syncthreads();
#else
    if (threadIdx.x < 192) w[threadIdx.x] = cw[(size_t)lo * 192 + threadIdx.x];
    __syncthreads();
#endif

    float acc = cb[lo];
    const float* base = xp + (size_t)b * 64 * 1024;
#pragma unroll 4
    for (int li = 0; li < 64; ++li) {
        const float* row = base + li * 1024;
        float xm = (d > 0)    ? row[d - 1] : 0.0f;
        float xc = row[d];
        float xq = (d < 1023) ? row[d + 1] : 0.0f;
        acc += xm * w[li * 3 + 0] + xc * w[li * 3 + 1] + xq * w[li * 3 + 2];
    }
    acc = acc / (1.0f + __expf(-acc));
    const size_t idx = (size_t)bo * 1024 + d;
    x_one[idx]    = acc;
    x_one_bf[idx] = f32_to_bf16(acc);
}

// Fused selective-scan + output gate. One wave32 per (b,e): lane owns states
// n = 4*lane + j (j=0..3) so B/C/A loads are single b128s; h stays in regs;
// never materializes (B,L,D,N).
__global__ __launch_bounds__(256) void ssm_scan_kernel(
    const float* __restrict__ delta, const float* __restrict__ dbc,
    const float* __restrict__ x_one, const float* __restrict__ x_two,
    const float* __restrict__ x_in,  const float* __restrict__ A_log,
    const float* __restrict__ Dp,    unsigned short* __restrict__ gated)
{
    const int wid  = (blockIdx.x * blockDim.x + threadIdx.x) >> 5;
    const int lane = threadIdx.x & 31;
    const int b  = wid >> 10;            // 8192 waves = 8 * 1024
    const int e  = wid & 1023;
    const int nb = lane << 2;            // this lane's 4 contiguous states

    const v4f Av = *(const v4f*)(A_log + (size_t)e * 128 + nb);
    float Ae[4], h[4];
#pragma unroll
    for (int j = 0; j < 4; ++j) {
        Ae[j] = -__expf(Av[j]);
        h[j]  = 0.0f;
    }
    const float dpe = Dp[e];

    for (int l = 0; l < 64; ++l) {
        const int row   = b * 64 + l;
        const size_t ei = (size_t)row * 1024 + e;
        const float dlt = delta[ei];
        const float xv  = x_one[ei];
        const float dx  = dlt * xv;
        const float* bc = dbc + (size_t)row * 320;   // [0:64)=draw [64:192)=B [192:320)=C
        const v4f Bm = *(const v4f*)(bc + 64 + nb);  // one b128
        const v4f Cm = *(const v4f*)(bc + 192 + nb); // one b128
        float part = 0.0f;
#pragma unroll
        for (int j = 0; j < 4; ++j) {
            const float dA = __expf(dlt * Ae[j]);
            h[j] = dA * h[j] + dx * Bm[j];
            part += h[j] * Cm[j];
        }
#pragma unroll
        for (int off = 16; off > 0; off >>= 1)       // wave32 butterfly reduce
            part += __shfl_xor(part, off, 32);
        if (lane == 0) {
            const float y = part + dpe * xv;
            const float g = y * x_two[ei] + x_in[ei];
            gated[ei] = f32_to_bf16(g);
        }
    }
}

extern "C" void kernel_launch(void* const* d_in, const int* in_sizes, int n_in,
                              void* d_out, int out_size, void* d_ws, size_t ws_size,
                              hipStream_t stream)
{
    (void)in_sizes; (void)n_in; (void)out_size; (void)ws_size;
    const float* x      = (const float*)d_in[0];
    const float* proj_w = (const float*)d_in[1];
    const float* proj_b = (const float*)d_in[2];
    const float* conv_w = (const float*)d_in[3];
    const float* conv_b = (const float*)d_in[4];
    const float* dbc_w  = (const float*)d_in[5];
    const float* dt_w   = (const float*)d_in[6];
    const float* dt_b   = (const float*)d_in[7];
    const float* A_log  = (const float*)d_in[8];
    const float* Dp     = (const float*)d_in[9];
    float* out = (float*)d_out;

    char* ws = (char*)d_ws;
    auto alloc = [&](size_t bytes) {
        char* p = ws; ws += (bytes + 255) & ~(size_t)255; return p;
    };
    unsigned short* x_bf     = (unsigned short*)alloc(512 * 1024 * 2);
    unsigned short* pw_bf    = (unsigned short*)alloc(1024 * 1024 * 2);
    unsigned short* dbcw_bf  = (unsigned short*)alloc(320 * 1024 * 2);
    unsigned short* dtw_bf   = (unsigned short*)alloc(1024 * 64 * 2);
    float*          xp       = (float*)alloc(512 * 1024 * 4);
    float*          x_two    = (float*)alloc(512 * 1024 * 4);
    float*          x_one    = (float*)alloc(512 * 1024 * 4);
    unsigned short* x_one_bf = (unsigned short*)alloc(512 * 1024 * 2);
    float*          dbc      = (float*)alloc(512 * 320 * 4);
    unsigned short* draw_bf  = (unsigned short*)alloc(512 * 64 * 2);
    float*          delta    = (float*)alloc(512 * 1024 * 4);
    unsigned short* gated_bf = (unsigned short*)alloc(512 * 1024 * 2);

    // 1) bf16 conversions for WMMA operands
    convert_f32_bf16<<<(512 * 1024 + 255) / 256, 256, 0, stream>>>(x, x_bf, 512 * 1024);
    convert_f32_bf16<<<(1024 * 1024 + 255) / 256, 256, 0, stream>>>(proj_w, pw_bf, 1024 * 1024);
    convert_f32_bf16<<<(320 * 1024 + 255) / 256, 256, 0, stream>>>(dbc_w, dbcw_bf, 320 * 1024);
    convert_f32_bf16<<<(1024 * 64 + 255) / 256, 256, 0, stream>>>(dt_w, dtw_bf, 1024 * 64);

    // 2) xp = x @ proj_w^T + b ; x_two = silu(xp)   [M=512,N=1024,K=1024 -> 64 blocks]
    gemm_wmma_bf16<MODE_XP_SILU><<<64, 256, 0, stream>>>(
        x_bf, 1024, pw_bf, 1024, proj_b, 512, 1024, 1024, xp, x_two, nullptr);

    // 3) x_one = silu(conv1d(xp)) (+ bf16 copy)
    conv_silu_kernel<<<dim3(512, 4), 256, 0, stream>>>(xp, conv_w, conv_b, x_one, x_one_bf);

    // 4) dbc = x_one @ dbc_w^T   [N=320 -> 160 wave tiles -> 20 blocks]
    gemm_wmma_bf16<MODE_DBC><<<20, 256, 0, stream>>>(
        x_one_bf, 1024, dbcw_bf, 1024, nullptr, 512, 320, 1024, dbc, nullptr, draw_bf);

    // 5) delta = softplus(draw @ dt_w^T + dt_b)   [K=64]
    gemm_wmma_bf16<MODE_SOFTPLUS><<<64, 256, 0, stream>>>(
        draw_bf, 64, dtw_bf, 64, dt_b, 512, 1024, 64, delta, nullptr, nullptr);

    // 6) fused scan + y*x_two + skip -> gated (bf16)   [8192 waves]
    ssm_scan_kernel<<<1024, 256, 0, stream>>>(
        delta, dbc, x_one, x_two, x, A_log, Dp, gated_bf);

    // 7) out = gated @ proj_w^T + proj_b
    gemm_wmma_bf16<MODE_PLAIN><<<64, 256, 0, stream>>>(
        gated_bf, 1024, pw_bf, 1024, proj_b, 512, 1024, 1024, out, nullptr, nullptr);
}
